// GraphAttention_57286273794661
// MI455X (gfx1250) — matrix operand order repaired and checked by reading, not compiled, not run on hardware
//
#include <hip/hip_runtime.h>

// ---------------------------------------------------------------------------
// GAT layer, fused flash-style, bf16 WMMA on gfx1250 (wave32)
// N=4096 nodes, F=512 in-features, U=64 per-head features, H=8 heads
// ---------------------------------------------------------------------------

#define Nn 4096
#define Ff 512
#define Uu 64
#define Hh 8
#define LRELU_SLOPE 0.2f
#define NEG_MASK -10e9f   // == -1e10

typedef __attribute__((ext_vector_type(16))) __bf16 v16bf;
typedef __attribute__((ext_vector_type(8)))  float  v8f;

__device__ __forceinline__ __bf16 f2bf(float f) {
    return static_cast<__bf16>(f);   // lowers to v_cvt_pk_bf16_f32 (paired)
}
__device__ __forceinline__ unsigned short f2bf_bits(float f) {
    __bf16 b = static_cast<__bf16>(f);
    unsigned short h; __builtin_memcpy(&h, &b, 2);
    return h;
}
__device__ __forceinline__ float bf2f(unsigned short h) {
    unsigned int u = ((unsigned int)h) << 16;
    float f; __builtin_memcpy(&f, &u, 4);
    return f;
}

// ---------------------------------------------------------------------------
// Kernel 1: per-head projection  hT[h][u][n] = sum_f X[n,f] * W[h,f,u]  (bf16)
// One wave per 16x16 output tile; X/W tiles staged in LDS with b128 loads.
// ---------------------------------------------------------------------------
__global__ __launch_bounds__(32) void gat_proj_kernel(
        const float* __restrict__ X,     // [N, F]
        const float* __restrict__ W,     // [H, F, U]
        unsigned short* __restrict__ hT) // [H, U, N] bf16 bits
{
    __shared__ float sX[16 * 32];   // X tile: [row n (16)][k (32)]
    __shared__ float sW[32 * 16];   // W tile: [k (32)][col u (16)]

    const int ntile = blockIdx.x;          // N/16
    const int utile = blockIdx.y;          // U/16 = 4
    const int head  = blockIdx.z;          // H
    const int lane  = threadIdx.x & 31;
    const int m     = lane & 15;
    const int half  = lane >> 4;
    const int n0 = ntile * 16;
    const int u0 = utile * 16;

    v8f acc = {};
    for (int k0 = 0; k0 < Ff; k0 += 32) {
        // ---- stage tiles (float4-granular, coalesced) ----
#pragma unroll
        for (int e = 0; e < 4; ++e) {
            const int idx = lane + e * 32;          // 0..127 float4 slots
            const int xr = idx >> 3, xc = idx & 7;  // 8 float4 per X row
            ((float4*)sX)[idx] =
                *(const float4*)&X[(size_t)(n0 + xr) * Ff + k0 + xc * 4];
            const int wr = idx >> 2, wc = idx & 3;  // 4 float4 per W row
            ((float4*)sW)[idx] =
                *(const float4*)&W[((size_t)head * Ff + k0 + wr) * Uu + u0 + wc * 4];
        }
        __syncthreads();

        // ---- A operand: lane row m, K runs [half*8..+7], [16+half*8..+7] ----
        v16bf a, b;
        {
            const float4* p0 = (const float4*)(sX + m * 32 + half * 8);
            const float4* p1 = (const float4*)(sX + m * 32 + 16 + half * 8);
            const float4 x0 = p0[0], x1 = p0[1], x2 = p1[0], x3 = p1[1];
            a[0] = f2bf(x0.x);  a[1] = f2bf(x0.y);  a[2]  = f2bf(x0.z);  a[3]  = f2bf(x0.w);
            a[4] = f2bf(x1.x);  a[5] = f2bf(x1.y);  a[6]  = f2bf(x1.z);  a[7]  = f2bf(x1.w);
            a[8] = f2bf(x2.x);  a[9] = f2bf(x2.y);  a[10] = f2bf(x2.z);  a[11] = f2bf(x2.w);
            a[12] = f2bf(x3.x); a[13] = f2bf(x3.y); a[14] = f2bf(x3.z);  a[15] = f2bf(x3.w);
        }
        // ---- B operand: lane col m, K runs [half*8..+7], [16+half*8..+7] ----
#pragma unroll
        for (int i = 0; i < 16; ++i)
            b[i] = f2bf(sW[((i >> 3) * 16 + half * 8 + (i & 7)) * 16 + m]);

        acc = __builtin_amdgcn_wmma_f32_16x16x32_bf16(
                false, a, false, b, (short)0, acc, false, false);
        __syncthreads();
    }
    // C/D layout: VGPR r -> row M = r + 8*half, col N = m
#pragma unroll
    for (int r = 0; r < 8; ++r) {
        const int M = r + 8 * half;
        hT[((size_t)head * Uu + (u0 + m)) * Nn + (n0 + M)] = f2bf_bits(acc[r]);
    }
}

// ---------------------------------------------------------------------------
// Kernel 1b: attention projections  f_self[h][n], f_neigh[h][n]
// ---------------------------------------------------------------------------
__global__ __launch_bounds__(256) void gat_fvec_kernel(
        const unsigned short* __restrict__ hT,      // [H, U, N]
        const float* __restrict__ a_self,           // [H, U]
        const float* __restrict__ a_neigh,          // [H, U]
        float* __restrict__ f_self,                 // [H, N]
        float* __restrict__ f_neigh)                // [H, N]
{
    const int idx  = blockIdx.x * blockDim.x + threadIdx.x;  // H*N
    const int head = idx >> 12;        // /4096
    const int n    = idx & (Nn - 1);
    float fs = 0.f, fn = 0.f;
#pragma unroll 8
    for (int u = 0; u < Uu; ++u) {
        const float hv = bf2f(hT[((size_t)head * Uu + u) * Nn + n]);
        fs += hv * a_self[head * Uu + u];
        fn += hv * a_neigh[head * Uu + u];
    }
    f_self[idx]  = fs;
    f_neigh[idx] = fn;
}

// ---------------------------------------------------------------------------
// Kernel 2: fused masked-softmax attention + aggregation + bias + relu
// Block = 256 threads = 8 waves; wave w handles head w for one 16-row tile.
// Double-buffered LDS staging: one barrier per j-step; next A tile loads
// overlap current tile's exp+WMMA work. Mask folded in at staging (1 FMA).
// Single-pass softmax: exp of masked score underflows to exactly 0.
// ---------------------------------------------------------------------------
__global__ __launch_bounds__(256) void gat_attn_kernel(
        const float* __restrict__ A,            // [N, N] dense adjacency
        const unsigned short* __restrict__ hT,  // [H, U, N] bf16 bits
        const float* __restrict__ f_self,       // [H, N]
        const float* __restrict__ f_neigh,      // [H, N]
        const float* __restrict__ bias,         // [H, U]
        float* __restrict__ out)                // [N, H*U]
{
    __shared__ float sA[2][16 * 32];   // pre-masked adjacency tiles (ping-pong)
    __shared__ float sFn[2][Hh * 32];  // per-head f_neigh slices (ping-pong)

    const int i0   = blockIdx.x * 16;
    const int tid  = threadIdx.x;
    const int head = tid >> 5;          // wave index == head
    const int lane = tid & 31;
    const int m    = lane & 15;
    const int half = lane >> 4;

    const float fs = f_self[(size_t)head * Nn + i0 + m];

    v8f acc0 = {}, acc1 = {}, acc2 = {}, acc3 = {};
    float lsum = 0.f;

    // stage(buf, jj): pre-masked A tile (128 x float4) + per-head f_neigh
#define GAT_STAGE(buf, jj)                                                    \
    {                                                                         \
        if (tid < 128) {                                                      \
            const int row = tid >> 3, c4 = tid & 7;                           \
            float4 v = *(const float4*)&A[(size_t)(i0 + row) * Nn + (jj) + c4 * 4]; \
            float4 t;                                                         \
            t.x = fmaf(v.x, -NEG_MASK, NEG_MASK);   /* NEG_MASK*(1-a) */      \
            t.y = fmaf(v.y, -NEG_MASK, NEG_MASK);                             \
            t.z = fmaf(v.z, -NEG_MASK, NEG_MASK);                             \
            t.w = fmaf(v.w, -NEG_MASK, NEG_MASK);                             \
            ((float4*)sA[buf])[tid] = t;                                      \
        }                                                                     \
        sFn[buf][head * 32 + lane] = f_neigh[(size_t)head * Nn + (jj) + lane];\
    }

    GAT_STAGE(0, 0)
    __syncthreads();

    for (int j0 = 0; j0 < Nn; j0 += 32) {
        const int cur = (j0 >> 5) & 1;
        if (j0 + 32 < Nn) {
            GAT_STAGE(1 - cur, j0 + 32)                 // overlap next tile
            __builtin_prefetch(A + (size_t)(i0 + m) * Nn + j0 + 64 + half * 16, 0, 1);
        }

        // ---- build P = exp(lrelu(fs+fn) + mask) in A-operand bf16 layout ----
        v16bf P;
        {
            const float4* pa0 = (const float4*)(sA[cur] + m * 32 + half * 8);
            const float4* pa1 = (const float4*)(sA[cur] + m * 32 + 16 + half * 8);
            const float4* pf0 = (const float4*)(sFn[cur] + head * 32 + half * 8);
            const float4* pf1 = (const float4*)(sFn[cur] + head * 32 + 16 + half * 8);
            const float4 A0 = pa0[0], A1 = pa0[1], A2 = pa1[0], A3 = pa1[1];
            const float4 F0 = pf0[0], F1 = pf0[1], F2 = pf1[0], F3 = pf1[1];
#define GAT_PE(i, fn_, msk_)                                                  \
            {                                                                 \
                float s = fs + (fn_);                                         \
                s = fmaxf(s, 0.f) + LRELU_SLOPE * fminf(s, 0.f);              \
                s += (msk_);                                                  \
                const float p = __expf(s);                                    \
                lsum += p;                                                    \
                P[i] = f2bf(p);                                               \
            }
            GAT_PE(0,  F0.x, A0.x) GAT_PE(1,  F0.y, A0.y)
            GAT_PE(2,  F0.z, A0.z) GAT_PE(3,  F0.w, A0.w)
            GAT_PE(4,  F1.x, A1.x) GAT_PE(5,  F1.y, A1.y)
            GAT_PE(6,  F1.z, A1.z) GAT_PE(7,  F1.w, A1.w)
            GAT_PE(8,  F2.x, A2.x) GAT_PE(9,  F2.y, A2.y)
            GAT_PE(10, F2.z, A2.z) GAT_PE(11, F2.w, A2.w)
            GAT_PE(12, F3.x, A3.x) GAT_PE(13, F3.y, A3.y)
            GAT_PE(14, F3.z, A3.z) GAT_PE(15, F3.w, A3.w)
#undef GAT_PE
        }

        // ---- 4 WMMAs: acc[16x16(u)] += P[16x32] @ hT-tile[32x16] ----
        const unsigned short* bbase =
            hT + (size_t)head * Uu * Nn + j0 + half * 8;  // B layout: K=half*8+i, 16+half*8+i
#define GAT_BTILE(t, ACC)                                                     \
        {                                                                     \
            union { uint4 q[2]; v16bf v; } B;                                 \
            const uint4* src =                                                \
                (const uint4*)(bbase + (size_t)((t) * 16 + m) * Nn);          \
            B.q[0] = src[0];                                                  \
        B.q[1] = src[2];                                                  \
            ACC = __builtin_amdgcn_wmma_f32_16x16x32_bf16(                    \
                    false, P, false, B.v, (short)0, ACC, false, false);       \
        }
        GAT_BTILE(0, acc0)
        GAT_BTILE(1, acc1)
        GAT_BTILE(2, acc2)
        GAT_BTILE(3, acc3)
#undef GAT_BTILE

        __syncthreads();   // staging of next buffer done; reads of cur done
    }
#undef GAT_STAGE

    // full row sums: lanes (m, m+16) hold disjoint K halves
    const float ltot = lsum + __shfl_xor(lsum, 16, 32);

    // ---- normalize, bias, relu, store: out[n, head*U + u] ----
#pragma unroll
    for (int r = 0; r < 8; ++r) {
        const int M = r + 8 * half;                 // row within tile
        const float linv = 1.0f / __shfl(ltot, M, 32);
        float* orow = out + (size_t)(i0 + M) * (Hh * Uu) + head * Uu + m;
        float v0 = acc0[r] * linv + bias[head * Uu +  0 + m];
        float v1 = acc1[r] * linv + bias[head * Uu + 16 + m];
        float v2 = acc2[r] * linv + bias[head * Uu + 32 + m];
        float v3 = acc3[r] * linv + bias[head * Uu + 48 + m];
        orow[ 0] = fmaxf(v0, 0.f);
        orow[16] = fmaxf(v1, 0.f);
        orow[32] = fmaxf(v2, 0.f);
        orow[48] = fmaxf(v3, 0.f);
    }
}

// ---------------------------------------------------------------------------
// Launch
// ---------------------------------------------------------------------------
extern "C" void kernel_launch(void* const* d_in, const int* in_sizes, int n_in,
                              void* d_out, int out_size, void* d_ws, size_t ws_size,
                              hipStream_t stream) {
    (void)in_sizes; (void)n_in; (void)out_size; (void)ws_size;
    const float* X       = (const float*)d_in[0];   // [1, N, F]
    // d_in[1] = out_indices (unused: reference returns all nodes)
    const float* A       = (const float*)d_in[2];   // [N, N]
    const float* W       = (const float*)d_in[3];   // [H, F, U]
    const float* a_self  = (const float*)d_in[4];   // [H, U]
    const float* a_neigh = (const float*)d_in[5];   // [H, U]
    const float* bias    = (const float*)d_in[6];   // [H, U]
    float* out = (float*)d_out;                     // [1, N, H*U]

    // workspace layout
    unsigned short* hT = (unsigned short*)d_ws;                       // 4 MB
    float* f_self  = (float*)((char*)d_ws + (size_t)Hh * Uu * Nn * 2);
    float* f_neigh = f_self + (size_t)Hh * Nn;

    gat_proj_kernel<<<dim3(Nn / 16, Uu / 16, Hh), 32, 0, stream>>>(X, W, hT);
    gat_fvec_kernel<<<(Hh * Nn) / 256, 256, 0, stream>>>(hT, a_self, a_neigh,
                                                         f_self, f_neigh);
    gat_attn_kernel<<<Nn / 16, 256, 0, stream>>>(A, hT, f_self, f_neigh, bias, out);
}
